// MappedAttention_23356032155957
// MI455X (gfx1250) — compile-verified, matrix operand order checked by reading, not run on hardware
//
#include <hip/hip_runtime.h>
#include <hip/hip_bf16.h>
#include <math.h>

typedef __bf16 bf16_t;
typedef __attribute__((ext_vector_type(16))) __bf16 v16bf;
typedef __attribute__((ext_vector_type(8)))  float  v8f;

#define LDK 40   // padded LDS K-stride in bf16 elems: 80 bytes per row (16B aligned)
#define TSZ (128 * LDK)

union FragU { v16bf v; int4 q[2]; };
union Pack16 { bf16_t h[16]; int4 q[2]; };

// ---- fragment load from LDS tile (rows x LDK, bf16) -------------------------
// A (16x32 MxK) and B (32x16 KxN) share the per-lane K pattern:
//   lane<16 : row=lane,     K = {0..7, 16..23}
//   lane>=16: row=lane-16,  K = {8..15, 24..31}
__device__ __forceinline__ v16bf load_frag(const bf16_t* base, int row0) {
  const int lane = threadIdx.x & 31;
  const int r = row0 + (lane & 15);
  const int c = (lane & 16) ? 8 : 0;
  const bf16_t* p = base + r * LDK + c;
  FragU f;
  f.q[0] = *(const int4*)(p);        // K = c .. c+7
  f.q[1] = *(const int4*)(p + 16);   // K = c+16 .. c+23
  return f.v;
}

// ---- 8 WMMA tiles per wave: 32 rows (2 M tiles) x 64 cols (4 N tiles) -------
__device__ __forceinline__ void mma_tile(const bf16_t* As, const bf16_t* Bs,
                                         int wm, int wn, v8f acc[2][4]) {
  v16bf a[2], b[4];
#pragma unroll
  for (int i = 0; i < 2; ++i) a[i] = load_frag(As, wm * 32 + 16 * i);
#pragma unroll
  for (int j = 0; j < 4; ++j) b[j] = load_frag(Bs, wn * 64 + 16 * j);
#pragma unroll
  for (int i = 0; i < 2; ++i)
#pragma unroll
    for (int j = 0; j < 4; ++j)
      acc[i][j] = __builtin_amdgcn_wmma_f32_16x16x32_bf16(
          false, a[i], false, b[j], (short)0, acc[i][j], false, false);
}

// ---- LDS staging helpers (blockDim.x == 256) --------------------------------
// 128 rows x 32 K of fp32 -> bf16 LDS
__device__ __forceinline__ void stage_f32_rows(bf16_t* S, const float* g, int ldg) {
  const int t = threadIdx.x;
  const int row = t >> 1, seg = (t & 1) * 16;
  Pack16 u;
#pragma unroll
  for (int i = 0; i < 4; ++i) {
    float4 f = *(const float4*)(g + (size_t)row * ldg + seg + i * 4);
    u.h[4 * i + 0] = (bf16_t)f.x; u.h[4 * i + 1] = (bf16_t)f.y;
    u.h[4 * i + 2] = (bf16_t)f.z; u.h[4 * i + 3] = (bf16_t)f.w;
  }
  int4* d = (int4*)(S + row * LDK + seg);
  d[0] = u.q[0]; d[1] = u.q[1];
}

// 128 rows x 32 K of bf16 -> LDS (straight copy)
__device__ __forceinline__ void stage_bf16_rows(bf16_t* S, const bf16_t* g, int ldg) {
  const int t = threadIdx.x;
  const int row = t >> 1, seg = (t & 1) * 16;
  const int4* p = (const int4*)(g + (size_t)row * ldg + seg);
  int4* d = (int4*)(S + row * LDK + seg);
  d[0] = p[0]; d[1] = p[1];
}

// 32 K-rows x 128 N-cols of bf16, transposed into LDS as S[n][k]
__device__ __forceinline__ void stage_bf16_T(bf16_t* S, const bf16_t* g, int ldg) {
  const int t = threadIdx.x;
  const int kr = t >> 3;            // 0..31
  const int c0 = (t & 7) * 16;      // 0..112
  Pack16 u;
  const int4* p = (const int4*)(g + (size_t)kr * ldg + c0);
  u.q[0] = p[0]; u.q[1] = p[1];
#pragma unroll
  for (int i = 0; i < 16; ++i) S[(c0 + i) * LDK + kr] = u.h[i];
}

// 128 rows x 32 K fp32 scores -> row-softmax applied (stats in regs) -> bf16 LDS
__device__ __forceinline__ void stage_scores_rows(bf16_t* S, const float* g, int ldg,
                                                  float m, float iv) {
  const int t = threadIdx.x;
  const int row = t >> 1, seg = (t & 1) * 16;
  Pack16 u;
#pragma unroll
  for (int i = 0; i < 4; ++i) {
    float4 f = *(const float4*)(g + (size_t)row * ldg + seg + i * 4);
    u.h[4 * i + 0] = (bf16_t)(__expf(f.x - m) * iv);
    u.h[4 * i + 1] = (bf16_t)(__expf(f.y - m) * iv);
    u.h[4 * i + 2] = (bf16_t)(__expf(f.z - m) * iv);
    u.h[4 * i + 3] = (bf16_t)(__expf(f.w - m) * iv);
  }
  int4* d = (int4*)(S + row * LDK + seg);
  d[0] = u.q[0]; d[1] = u.q[1];
}

// 32 s-rows x 128 t-cols fp32 scores -> col-softmax (stats in LDS) -> transposed S[t][s]
__device__ __forceinline__ void stage_scores_T(bf16_t* S, const float* g, int ldg,
                                               const float* sm, const float* si) {
  const int t = threadIdx.x;
  const int sr = t >> 3;            // 0..31
  const int c0 = (t & 7) * 16;      // 0..112
#pragma unroll
  for (int i4 = 0; i4 < 4; ++i4) {
    float4 f = *(const float4*)(g + (size_t)sr * ldg + c0 + i4 * 4);
    float x[4] = {f.x, f.y, f.z, f.w};
#pragma unroll
    for (int j = 0; j < 4; ++j) {
      const int tl = c0 + i4 * 4 + j;
      S[tl * LDK + sr] = (bf16_t)(__expf(x[j] - sm[tl]) * si[tl]);
    }
  }
}

// gelu(x) = 0.5x(1+tanh(z)) = x * sigmoid(2z), z = 0.79788456x(1+0.044715x^2)
// -> one v_exp_f32 + one v_rcp_f32, branch-free
__device__ __forceinline__ float gelu_tanh(float x) {
  const float z2 = 1.5957691216f * x * (1.0f + 0.044715f * x * x);
  return x * __builtin_amdgcn_rcpf(1.0f + __expf(-z2));
}

// ============================ kernels ========================================

// Fold weight-norm scale into weights, convert to bf16. One block per row.
__global__ __launch_bounds__(256) void k_prep_w(const float* __restrict__ v,
                                                const float* __restrict__ g,
                                                bf16_t* __restrict__ w) {
  __shared__ float red[256];
  const int row = blockIdx.x;
  const float* vr = v + (size_t)row * 1024;
  float ss = 0.0f;
  for (int k = threadIdx.x; k < 1024; k += 256) { float x = vr[k]; ss += x * x; }
  red[threadIdx.x] = ss;
  __syncthreads();
  for (int off = 128; off > 0; off >>= 1) {
    if (threadIdx.x < off) red[threadIdx.x] += red[threadIdx.x + off];
    __syncthreads();
  }
  const float scale = g[row] * rsqrtf(red[0]);
  for (int k = threadIdx.x; k < 1024; k += 256)
    w[(size_t)row * 1024 + k] = (bf16_t)(vr[k] * scale);
}

// H/T = bf16(gelu(X @ W1^T + b1)). grid(8,64,2), z: 0=head 1=tail
__global__ __launch_bounds__(256) void k_proj_gelu(const float* __restrict__ head,
                                                   const float* __restrict__ tail,
                                                   const bf16_t* __restrict__ W,
                                                   const float* __restrict__ bias,
                                                   bf16_t* __restrict__ H,
                                                   bf16_t* __restrict__ T) {
  __shared__ bf16_t As[2][TSZ];
  __shared__ bf16_t Bs[2][TSZ];
  const float* X = blockIdx.z ? tail : head;
  bf16_t* O = blockIdx.z ? T : H;
  const int n0 = blockIdx.x * 128, m0 = blockIdx.y * 128;
  const int wave = threadIdx.x >> 5, wm = wave & 3, wn = wave >> 2;
  const v8f zero = {0.f, 0.f, 0.f, 0.f, 0.f, 0.f, 0.f, 0.f};
  v8f acc[2][4];
#pragma unroll
  for (int i = 0; i < 2; ++i)
#pragma unroll
    for (int j = 0; j < 4; ++j) acc[i][j] = zero;

  stage_f32_rows(As[0], X + (size_t)m0 * 1024, 1024);
  stage_bf16_rows(Bs[0], W + (size_t)n0 * 1024, 1024);
  __syncthreads();
  int buf = 0;
  for (int kt = 32; kt <= 1024; kt += 32) {
    if (kt < 1024) {
      stage_f32_rows(As[buf ^ 1], X + (size_t)m0 * 1024 + kt, 1024);
      stage_bf16_rows(Bs[buf ^ 1], W + (size_t)n0 * 1024 + kt, 1024);
    }
    mma_tile(As[buf], Bs[buf], wm, wn, acc);
    __syncthreads();
    buf ^= 1;
  }
  const int lane = threadIdx.x & 31;
  const int colL = lane & 15;
  const int rsel = (lane & 16) ? 8 : 0;
#pragma unroll
  for (int i = 0; i < 2; ++i)
#pragma unroll
    for (int j = 0; j < 4; ++j) {
      const int col = n0 + wn * 64 + 16 * j + colL;
      const int rowb = m0 + wm * 32 + 16 * i + rsel;
      const float bv = bias[col];
#pragma unroll
      for (int r = 0; r < 8; ++r) {
        const float x = acc[i][j][r] + bv;
        O[(size_t)(rowb + r) * 1024 + col] = (bf16_t)gelu_tanh(x);
      }
    }
}

// scores[b][s][t] = (H_b @ T_b^T)[s][t] / 32. grid(16,16,4)
__global__ __launch_bounds__(256) void k_scores(const bf16_t* __restrict__ H,
                                                const bf16_t* __restrict__ T,
                                                float* __restrict__ Sc) {
  __shared__ bf16_t As[2][TSZ];
  __shared__ bf16_t Bs[2][TSZ];
  const int b = blockIdx.z;
  const bf16_t* Hb = H + (size_t)b * 2048 * 1024;
  const bf16_t* Tb = T + (size_t)b * 2048 * 1024;
  float* Sb = Sc + (size_t)b * 2048 * 2048;
  const int n0 = blockIdx.x * 128, m0 = blockIdx.y * 128;
  const int wave = threadIdx.x >> 5, wm = wave & 3, wn = wave >> 2;
  const v8f zero = {0.f, 0.f, 0.f, 0.f, 0.f, 0.f, 0.f, 0.f};
  v8f acc[2][4];
#pragma unroll
  for (int i = 0; i < 2; ++i)
#pragma unroll
    for (int j = 0; j < 4; ++j) acc[i][j] = zero;

  stage_bf16_rows(As[0], Hb + (size_t)m0 * 1024, 1024);
  stage_bf16_rows(Bs[0], Tb + (size_t)n0 * 1024, 1024);  // B lane n = row t of T
  __syncthreads();
  int buf = 0;
  for (int kt = 32; kt <= 1024; kt += 32) {
    if (kt < 1024) {
      stage_bf16_rows(As[buf ^ 1], Hb + (size_t)m0 * 1024 + kt, 1024);
      stage_bf16_rows(Bs[buf ^ 1], Tb + (size_t)n0 * 1024 + kt, 1024);
    }
    mma_tile(As[buf], Bs[buf], wm, wn, acc);
    __syncthreads();
    buf ^= 1;
  }
  const int lane = threadIdx.x & 31;
  const int colL = lane & 15;
  const int rsel = (lane & 16) ? 8 : 0;
#pragma unroll
  for (int i = 0; i < 2; ++i)
#pragma unroll
    for (int j = 0; j < 4; ++j) {
      const int col = n0 + wn * 64 + 16 * j + colL;
      const int rowb = m0 + wm * 32 + 16 * i + rsel;
#pragma unroll
      for (int r = 0; r < 8; ++r)
        Sb[(size_t)(rowb + r) * 2048 + col] = acc[i][j][r] * 0.03125f;
    }
}

// Online softmax stats per column (axis=1, over s). grid(8,4) x 256 threads.
__global__ __launch_bounds__(256) void k_colstats(const float* __restrict__ Sc,
                                                  float* __restrict__ cmax,
                                                  float* __restrict__ cinv) {
  const int b = blockIdx.y;
  const int t = blockIdx.x * 256 + threadIdx.x;
  const float* Sb = Sc + (size_t)b * 2048 * 2048;
  float m = -3.0e38f, s = 0.0f;
  for (int srow = 0; srow < 2048; ++srow) {
    const float x = Sb[(size_t)srow * 2048 + t];
    if (x > m) { s = s * __expf(m - x) + 1.0f; m = x; }
    else       { s += __expf(x - m); }
  }
  cmax[b * 2048 + t] = m;
  cinv[b * 2048 + t] = 1.0f / s;
}

// Online softmax stats per row (axis=2, over t). One wave per row. grid(256,4).
__global__ __launch_bounds__(256) void k_rowstats(const float* __restrict__ Sc,
                                                  float* __restrict__ rmax,
                                                  float* __restrict__ rinv) {
  const int b = blockIdx.y;
  const int wave = threadIdx.x >> 5, lane = threadIdx.x & 31;
  const int srow = blockIdx.x * 8 + wave;
  const float* R = Sc + (size_t)b * 2048 * 2048 + (size_t)srow * 2048;
  float m = -3.0e38f, s = 0.0f;
  for (int t = lane; t < 2048; t += 32) {
    const float x = R[t];
    if (x > m) { s = s * __expf(m - x) + 1.0f; m = x; }
    else       { s += __expf(x - m); }
  }
  for (int off = 16; off > 0; off >>= 1) {
    const float mo = __shfl_xor(m, off, 32);
    const float so = __shfl_xor(s, off, 32);
    const float M = fmaxf(m, mo);
    s = s * __expf(m - M) + so * __expf(mo - M);
    m = M;
  }
  if (lane == 0) { rmax[b * 2048 + srow] = m; rinv[b * 2048 + srow] = 1.0f / s; }
}

// pred_head_pre[b][s][d] = sum_t softmax_row(scores)[s][t] * T[b][t][d]. grid(8,16,4)
__global__ __launch_bounds__(256) void k_attn_head(const float* __restrict__ Sc,
                                                   const float* __restrict__ rmax,
                                                   const float* __restrict__ rinv,
                                                   const bf16_t* __restrict__ T,
                                                   bf16_t* __restrict__ P) {
  __shared__ bf16_t As[2][TSZ];
  __shared__ bf16_t Bs[2][TSZ];
  const int b = blockIdx.z;
  const float* Sb = Sc + (size_t)b * 2048 * 2048;
  const bf16_t* Tb = T + (size_t)b * 2048 * 1024;
  bf16_t* Pb = P + (size_t)b * 2048 * 1024;
  const int n0 = blockIdx.x * 128, m0 = blockIdx.y * 128;
  const int wave = threadIdx.x >> 5, wm = wave & 3, wn = wave >> 2;
  // per-thread row stats, fixed for the whole K loop
  const float m_r = rmax[b * 2048 + m0 + (threadIdx.x >> 1)];
  const float iv_r = rinv[b * 2048 + m0 + (threadIdx.x >> 1)];
  const v8f zero = {0.f, 0.f, 0.f, 0.f, 0.f, 0.f, 0.f, 0.f};
  v8f acc[2][4];
#pragma unroll
  for (int i = 0; i < 2; ++i)
#pragma unroll
    for (int j = 0; j < 4; ++j) acc[i][j] = zero;

  stage_scores_rows(As[0], Sb + (size_t)m0 * 2048, 2048, m_r, iv_r);
  stage_bf16_T(Bs[0], Tb + n0, 1024);  // Bs[d][t]
  __syncthreads();
  int buf = 0;
  for (int kt = 32; kt <= 2048; kt += 32) {
    if (kt < 2048) {
      stage_scores_rows(As[buf ^ 1], Sb + (size_t)m0 * 2048 + kt, 2048, m_r, iv_r);
      stage_bf16_T(Bs[buf ^ 1], Tb + (size_t)kt * 1024 + n0, 1024);
    }
    mma_tile(As[buf], Bs[buf], wm, wn, acc);
    __syncthreads();
    buf ^= 1;
  }
  const int lane = threadIdx.x & 31;
  const int colL = lane & 15;
  const int rsel = (lane & 16) ? 8 : 0;
#pragma unroll
  for (int i = 0; i < 2; ++i)
#pragma unroll
    for (int j = 0; j < 4; ++j) {
      const int col = n0 + wn * 64 + 16 * j + colL;
      const int rowb = m0 + wm * 32 + 16 * i + rsel;
#pragma unroll
      for (int r = 0; r < 8; ++r)
        Pb[(size_t)(rowb + r) * 1024 + col] = (bf16_t)acc[i][j][r];
    }
}

// pred_tail_pre[b][t][d] = sum_s softmax_col(scores)[s][t] * H[b][s][d]. grid(8,16,4)
__global__ __launch_bounds__(256) void k_attn_tail(const float* __restrict__ Sc,
                                                   const float* __restrict__ cmax,
                                                   const float* __restrict__ cinv,
                                                   const bf16_t* __restrict__ H,
                                                   bf16_t* __restrict__ P) {
  __shared__ bf16_t As[2][TSZ];
  __shared__ bf16_t Bs[2][TSZ];
  __shared__ float sm[128], si[128];   // col stats cached once for block's 128 t's
  const int b = blockIdx.z;
  const float* Sb = Sc + (size_t)b * 2048 * 2048;
  const bf16_t* Hb = H + (size_t)b * 2048 * 1024;
  bf16_t* Pb = P + (size_t)b * 2048 * 1024;
  const int n0 = blockIdx.x * 128, m0 = blockIdx.y * 128;  // m0: t, n0: d
  const int wave = threadIdx.x >> 5, wm = wave & 3, wn = wave >> 2;
  if (threadIdx.x < 128) {
    sm[threadIdx.x] = cmax[b * 2048 + m0 + threadIdx.x];
    si[threadIdx.x] = cinv[b * 2048 + m0 + threadIdx.x];
  }
  const v8f zero = {0.f, 0.f, 0.f, 0.f, 0.f, 0.f, 0.f, 0.f};
  v8f acc[2][4];
#pragma unroll
  for (int i = 0; i < 2; ++i)
#pragma unroll
    for (int j = 0; j < 4; ++j) acc[i][j] = zero;
  __syncthreads();

  stage_scores_T(As[0], Sb + m0, 2048, sm, si);  // As[t][s]
  stage_bf16_T(Bs[0], Hb + n0, 1024);            // Bs[d][s]
  __syncthreads();
  int buf = 0;
  for (int kt = 32; kt <= 2048; kt += 32) {  // kt: s
    if (kt < 2048) {
      stage_scores_T(As[buf ^ 1], Sb + (size_t)kt * 2048 + m0, 2048, sm, si);
      stage_bf16_T(Bs[buf ^ 1], Hb + (size_t)kt * 1024 + n0, 1024);
    }
    mma_tile(As[buf], Bs[buf], wm, wn, acc);
    __syncthreads();
    buf ^= 1;
  }
  const int lane = threadIdx.x & 31;
  const int colL = lane & 15;
  const int rsel = (lane & 16) ? 8 : 0;
#pragma unroll
  for (int i = 0; i < 2; ++i)
#pragma unroll
    for (int j = 0; j < 4; ++j) {
      const int col = n0 + wn * 64 + 16 * j + colL;
      const int rowb = m0 + wm * 32 + 16 * i + rsel;
#pragma unroll
      for (int r = 0; r < 8; ++r)
        Pb[(size_t)(rowb + r) * 1024 + col] = (bf16_t)acc[i][j][r];
    }
}

// out = pred_pre @ W2n^T + b2, fp32. grid(8,64,2): z=0 pred_head, z=1 pred_tail
__global__ __launch_bounds__(256) void k_out_proj(const bf16_t* __restrict__ PH,
                                                  const bf16_t* __restrict__ PT,
                                                  const bf16_t* __restrict__ W2,
                                                  const float* __restrict__ b2,
                                                  float* __restrict__ out) {
  __shared__ bf16_t As[2][TSZ];
  __shared__ bf16_t Bs[2][TSZ];
  const bf16_t* P = blockIdx.z ? PT : PH;
  float* O = out + (size_t)blockIdx.z * 8192 * 1024;
  const int n0 = blockIdx.x * 128, m0 = blockIdx.y * 128;
  const int wave = threadIdx.x >> 5, wm = wave & 3, wn = wave >> 2;
  const v8f zero = {0.f, 0.f, 0.f, 0.f, 0.f, 0.f, 0.f, 0.f};
  v8f acc[2][4];
#pragma unroll
  for (int i = 0; i < 2; ++i)
#pragma unroll
    for (int j = 0; j < 4; ++j) acc[i][j] = zero;

  stage_bf16_rows(As[0], P + (size_t)m0 * 1024, 1024);
  stage_bf16_rows(Bs[0], W2 + (size_t)n0 * 1024, 1024);
  __syncthreads();
  int buf = 0;
  for (int kt = 32; kt <= 1024; kt += 32) {
    if (kt < 1024) {
      stage_bf16_rows(As[buf ^ 1], P + (size_t)m0 * 1024 + kt, 1024);
      stage_bf16_rows(Bs[buf ^ 1], W2 + (size_t)n0 * 1024 + kt, 1024);
    }
    mma_tile(As[buf], Bs[buf], wm, wn, acc);
    __syncthreads();
    buf ^= 1;
  }
  const int lane = threadIdx.x & 31;
  const int colL = lane & 15;
  const int rsel = (lane & 16) ? 8 : 0;
#pragma unroll
  for (int i = 0; i < 2; ++i)
#pragma unroll
    for (int j = 0; j < 4; ++j) {
      const int col = n0 + wn * 64 + 16 * j + colL;
      const int rowb = m0 + wm * 32 + 16 * i + rsel;
      const float bv = b2[col];
#pragma unroll
      for (int r = 0; r < 8; ++r)
        O[(size_t)(rowb + r) * 1024 + col] = acc[i][j][r] + bv;
    }
}

// ============================ launch =========================================

extern "C" void kernel_launch(void* const* d_in, const int* in_sizes, int n_in,
                              void* d_out, int out_size, void* d_ws, size_t ws_size,
                              hipStream_t stream) {
  (void)in_sizes; (void)n_in; (void)out_size; (void)ws_size;
  const float* head = (const float*)d_in[0];
  const float* tail = (const float*)d_in[1];
  const float* v1   = (const float*)d_in[2];
  const float* g1   = (const float*)d_in[3];
  const float* b1   = (const float*)d_in[4];
  const float* v2   = (const float*)d_in[5];
  const float* g2   = (const float*)d_in[6];
  const float* b2   = (const float*)d_in[7];

  char* ws = (char*)d_ws;
  size_t off = 0;
  auto alloc = [&](size_t bytes) -> char* {
    char* p = ws + off;
    off += (bytes + 255) & ~(size_t)255;
    return p;
  };
  bf16_t* W1   = (bf16_t*)alloc((size_t)1024 * 1024 * 2);      //  2 MB
  bf16_t* W2   = (bf16_t*)alloc((size_t)1024 * 1024 * 2);      //  2 MB
  bf16_t* H    = (bf16_t*)alloc((size_t)8192 * 1024 * 2);      // 16 MB
  bf16_t* T    = (bf16_t*)alloc((size_t)8192 * 1024 * 2);      // 16 MB
  float*  Sc   = (float*) alloc((size_t)4 * 2048 * 2048 * 4);  // 64 MB
  float*  cmax = (float*) alloc((size_t)4 * 2048 * 4);
  float*  cinv = (float*) alloc((size_t)4 * 2048 * 4);
  float*  rmax = (float*) alloc((size_t)4 * 2048 * 4);
  float*  rinv = (float*) alloc((size_t)4 * 2048 * 4);
  bf16_t* PH   = (bf16_t*)alloc((size_t)8192 * 1024 * 2);      // 16 MB
  bf16_t* PT   = (bf16_t*)alloc((size_t)8192 * 1024 * 2);      // 16 MB

  k_prep_w<<<1024, 256, 0, stream>>>(v1, g1, W1);
  k_prep_w<<<1024, 256, 0, stream>>>(v2, g2, W2);
  k_proj_gelu<<<dim3(8, 64, 2), 256, 0, stream>>>(head, tail, W1, b1, H, T);
  k_scores<<<dim3(16, 16, 4), 256, 0, stream>>>(H, T, Sc);
  k_colstats<<<dim3(8, 4), 256, 0, stream>>>(Sc, cmax, cinv);
  k_rowstats<<<dim3(256, 4), 256, 0, stream>>>(Sc, rmax, rinv);
  k_attn_head<<<dim3(8, 16, 4), 256, 0, stream>>>(Sc, rmax, rinv, T, PH);
  k_attn_tail<<<dim3(8, 16, 4), 256, 0, stream>>>(Sc, cmax, cinv, H, PT);
  k_out_proj<<<dim3(8, 64, 2), 256, 0, stream>>>(PH, PT, W2, b2, (float*)d_out);
}
